// GatedTinyMambaLayer_39900246180109
// MI455X (gfx1250) — matrix operand order repaired and checked by reading, not compiled, not run on hardware
//
#include <hip/hip_runtime.h>
#include <cstdint>

typedef __attribute__((ext_vector_type(16))) __bf16 v16bf;
typedef __attribute__((ext_vector_type(8)))  __bf16 v8bf;
typedef __attribute__((ext_vector_type(8)))  float  v8f;

#define B_   8
#define T_   2048
#define DM_  1024
#define DS_  256
#define BT_  (B_ * T_)

__device__ __forceinline__ float sigm(float x) { return 1.0f / (1.0f + __expf(-x)); }

// A-fragment (16x32 bf16) from an LDS row-major tile, per ISA 16-bit A layout:
// lanes 0-15: M=lane, K = {0..7} then {16..23}; lanes 16-31: +8 on K.
template<int ROWLEN>
__device__ __forceinline__ v16bf load_a_frag(const __bf16* tile, int lane, int k0) {
    int m = lane & 15;
    int g = lane >> 4;
    const __bf16* r = tile + m * ROWLEN + k0 + 8 * g;
    v8bf lo = *(const v8bf*)(r);
    v8bf hi = *(const v8bf*)(r + 16);
    return __builtin_shufflevector(lo, hi, 0,1,2,3,4,5,6,7,8,9,10,11,12,13,14,15);
}

// B-fragment (32x16 bf16) from row-major bf16 weight W[N][K]:
// lane holds column n (n already includes lane&15); K = {0..7} then {16..23}; lanes 16-31: +8 on K.
__device__ __forceinline__ v16bf load_b_frag(const __bf16* W, int ldw, int n, int k0, int lane) {
    int g = lane >> 4;
    const __bf16* r = W + (size_t)n * ldw + k0 + 8 * g;
    v8bf lo = *(const v8bf*)(r);
    v8bf hi = *(const v8bf*)(r + 16);
    return __builtin_shufflevector(lo, hi, 0,1,2,3,4,5,6,7,8,9,10,11,12,13,14,15);
}

__device__ __forceinline__ v8f wmma_bf16(v16bf a, v16bf b, v8f c) {
    return __builtin_amdgcn_wmma_f32_16x16x32_bf16(false, a, false, b, (short)0, c, false, false);
}

// ---------------------------------------------------------------------------
// prepW: pack bf16 weights. Wcat[768][1024] = [W_in+W_x ; W_gx ; W_dc[:, :1024]]
//        Wppa[256][256] = W_pp[:, :256] ; Woutb[1024][256] = W_out
// ---------------------------------------------------------------------------
__global__ void __launch_bounds__(256) prepW(
    const float* __restrict__ W_in, const float* __restrict__ W_x,
    const float* __restrict__ W_gx, const float* __restrict__ W_dc,
    const float* __restrict__ W_pp, const float* __restrict__ W_out,
    __bf16* __restrict__ Wcat, __bf16* __restrict__ Wppa, __bf16* __restrict__ Woutb)
{
    const int NC = 768 * 1024;
    const int NP = 256 * 256;
    const int NO = 1024 * 256;
    int idx = blockIdx.x * 256 + threadIdx.x;
    if (idx < NC) {
        int r = idx >> 10, k = idx & 1023;
        float v;
        if (r < 256)      v = W_in[r * 1024 + k] + W_x[r * 1024 + k];
        else if (r < 512) v = W_gx[(r - 256) * 1024 + k];
        else              v = W_dc[(r - 512) * 1280 + k];
        Wcat[idx] = (__bf16)v;
    } else if (idx < NC + NP) {
        int j = idx - NC;
        int r = j >> 8, k = j & 255;
        Wppa[j] = (__bf16)W_pp[r * 512 + k];
    } else if (idx < NC + NP + NO) {
        int j = idx - NC - NP;
        Woutb[j] = (__bf16)W_out[j];
    }
}

// ---------------------------------------------------------------------------
// prepS: per-batch (8 rows) state projections, f32 VALU (tiny).
// ---------------------------------------------------------------------------
__global__ void __launch_bounds__(256) prepS(
    const float* __restrict__ prev,
    const float* __restrict__ W_s,
    const float* __restrict__ W_dc, const float* __restrict__ b_dc,
    const float* __restrict__ W_pp, const float* __restrict__ b_pp,
    const float* __restrict__ W_pg, const float* __restrict__ b_pg,
    const float* __restrict__ W_gs, const float* __restrict__ b_gs,
    float* __restrict__ sWs, float* __restrict__ dcs, float* __restrict__ pps,
    float* __restrict__ pstr, float* __restrict__ gstate)
{
    int bi  = blockIdx.x & 7;
    int mat = blockIdx.x >> 3;
    int c   = threadIdx.x;
    const float* s = prev + bi * DS_;
    float acc = 0.0f;
    if (mat == 0) {
        const float* w = W_s + c * DS_;
        #pragma unroll 4
        for (int j = 0; j < DS_; ++j) acc += s[j] * w[j];
        sWs[bi * DS_ + c] = 0.5f * acc;
    } else if (mat == 1) {
        const float* w = W_dc + c * 1280 + 1024;
        #pragma unroll 4
        for (int j = 0; j < DS_; ++j) acc += s[j] * w[j];
        dcs[bi * DS_ + c] = acc + b_dc[c];
    } else if (mat == 2) {
        const float* w = W_pp + c * 512 + 256;
        #pragma unroll 4
        for (int j = 0; j < DS_; ++j) acc += s[j] * w[j];
        pps[bi * DS_ + c] = acc + b_pp[c];
    } else if (mat == 3) {
        const float* w = W_pg + c * DS_;
        #pragma unroll 4
        for (int j = 0; j < DS_; ++j) acc += s[j] * w[j];
        pstr[bi * DS_ + c] = sigm(acc + b_pg[c]);
    } else {
        const float* w = W_gs + c * DS_;
        #pragma unroll 4
        for (int j = 0; j < DS_; ++j) acc += s[j] * w[j];
        gstate[bi * DS_ + c] = sigm(acc + b_gs[c]);
    }
}

// ---------------------------------------------------------------------------
// stageA: Y = x @ Wcat^T. Block = 32 tokens x 768 cols, 512 threads (16 waves).
// Wave w owns columns [16w,16w+16) of ALL THREE groups -> epilogue fully in
// registers (no LDS exchange). A staged 32x64 at a time, double-buffered.
// ---------------------------------------------------------------------------
__global__ void __launch_bounds__(512) stageA(
    const float* __restrict__ x, const float* __restrict__ prev,
    const __bf16* __restrict__ Wcat, const float* __restrict__ b_gx,
    const float* __restrict__ sWs, const float* __restrict__ dcs,
    float* __restrict__ s_pred, float* __restrict__ gate_in)
{
    __shared__ __bf16 As[2][32 * 64];
    const int tid  = threadIdx.x;
    const int lane = tid & 31;
    const int wv   = tid >> 5;        // 0..15
    const int ln   = lane & 15;
    const int g    = lane >> 4;
    const int Mbase = blockIdx.x * 32;

    // cooperative staging indices: 32 rows x 64 cols, float4 per thread
    const int lm = tid >> 4;          // 0..31
    const int lc = (tid & 15) * 4;    // 0..60
    const float* xp = x + (size_t)(Mbase + lm) * DM_ + lc;

    {   // prologue: stage tile 0
        float4 v = *(const float4*)(xp);
        __bf16* d = &As[0][lm * 64 + lc];
        d[0] = (__bf16)v.x; d[1] = (__bf16)v.y; d[2] = (__bf16)v.z; d[3] = (__bf16)v.w;
    }
    __syncthreads();

    v8f acc[2][3] = {};
    for (int it = 0; it < 16; ++it) {
        const int cur = it & 1;
        float4 nxt;
        if (it < 15) nxt = *(const float4*)(xp + (it + 1) * 64);

        // all B fragments for this 64-wide K slab: one clause, one wait
        v16bf bfr[2][3];
        #pragma unroll
        for (int sub = 0; sub < 2; ++sub) {
            int k0 = it * 64 + sub * 32;
            #pragma unroll
            for (int t = 0; t < 3; ++t) {
                int n = t * 256 + wv * 16 + ln;
                bfr[sub][t] = load_b_frag(Wcat, DM_, n, k0, lane);
            }
        }
        #pragma unroll
        for (int sub = 0; sub < 2; ++sub) {
            v16bf a0 = load_a_frag<64>(&As[cur][0],       lane, sub * 32);
            v16bf a1 = load_a_frag<64>(&As[cur][16 * 64], lane, sub * 32);
            #pragma unroll
            for (int t = 0; t < 3; ++t) {
                acc[0][t] = wmma_bf16(a0, bfr[sub][t], acc[0][t]);
                acc[1][t] = wmma_bf16(a1, bfr[sub][t], acc[1][t]);
            }
        }
        if (it < 15) {  // stage next slab into the other buffer
            __bf16* d = &As[cur ^ 1][lm * 64 + lc];
            d[0] = (__bf16)nxt.x; d[1] = (__bf16)nxt.y; d[2] = (__bf16)nxt.z; d[3] = (__bf16)nxt.w;
        }
        __syncthreads();
    }

    // register-resident epilogue: this wave holds tsum/tgx/tdcx for column c
    const int bi = blockIdx.x >> 6;     // 64 blocks of 32 tokens per batch
    const int c  = wv * 16 + ln;
    const float ps  = prev[bi * DS_ + c];
    const float sws = sWs[bi * DS_ + c];
    const float dcv = dcs[bi * DS_ + c];
    const float bgx = b_gx[c];
    #pragma unroll
    for (int ms = 0; ms < 2; ++ms) {
        #pragma unroll
        for (int v = 0; v < 8; ++v) {
            int m = ms * 16 + v + 8 * g;
            float tsum = acc[ms][0][v];
            float tgx  = acc[ms][1][v];
            float tdcx = acc[ms][2][v];
            float sg = sigm(tdcx + dcv);
            sg = fminf(fmaxf(sg, 0.0f), 1.0f);
            float decay = __expf(-0.05f * sg);
            size_t o = (size_t)(Mbase + m) * DS_ + c;
            s_pred[o]  = ps * decay + tsum + sws;
            gate_in[o] = sigm(tgx + bgx);
        }
    }
}

// ---------------------------------------------------------------------------
// stageB: pv = s_pred @ Wppa^T, fused tanh/perturb/gating epilogue -> h (bf16)
// and the h[:, -1, :] output tail.  16 tokens x 256 cols per 256-thread block.
// ---------------------------------------------------------------------------
__global__ void __launch_bounds__(256) stageB(
    const float* __restrict__ s_pred, const float* __restrict__ gate_in,
    const __bf16* __restrict__ Wppa,
    const float* __restrict__ pps, const float* __restrict__ pstr,
    const float* __restrict__ gstate,
    __bf16* __restrict__ h_bf, float* __restrict__ out_tail)
{
    __shared__ float  Sf[16 * 256];
    __shared__ __bf16 Sb[16 * 256];
    const int tid  = threadIdx.x;
    const int lane = tid & 31;
    const int wv   = tid >> 5;
    const int ln   = lane & 15;
    const int g    = lane >> 4;
    const int Mbase = blockIdx.x * 16;

    {   // cooperative load of 16x256 s_pred tile (f32 + bf16 copies)
        const float4* src = (const float4*)(s_pred + (size_t)Mbase * DS_);
        #pragma unroll
        for (int i = 0; i < 4; ++i) {
            float4 v = src[tid * 4 + i];
            int base = tid * 16 + i * 4;
            Sf[base + 0] = v.x; Sf[base + 1] = v.y; Sf[base + 2] = v.z; Sf[base + 3] = v.w;
            Sb[base + 0] = (__bf16)v.x; Sb[base + 1] = (__bf16)v.y;
            Sb[base + 2] = (__bf16)v.z; Sb[base + 3] = (__bf16)v.w;
        }
    }
    __syncthreads();

    v8f acc[2] = {};
    #pragma unroll 2
    for (int k0 = 0; k0 < DS_; k0 += 32) {
        v16bf bfr[2];
        #pragma unroll
        for (int t = 0; t < 2; ++t) {
            int n = wv * 32 + t * 16 + ln;
            bfr[t] = load_b_frag(Wppa, DS_, n, k0, lane);
        }
        v16bf a = load_a_frag<256>(Sb, lane, k0);
        #pragma unroll
        for (int t = 0; t < 2; ++t) acc[t] = wmma_bf16(a, bfr[t], acc[t]);
    }

    const int bi = blockIdx.x >> 7;
    #pragma unroll
    for (int t = 0; t < 2; ++t) {
        int n = wv * 32 + t * 16 + ln;
        float ppsv = pps[bi * DS_ + n];
        float pstv = pstr[bi * DS_ + n];
        float gst  = gstate[bi * DS_ + n];
        #pragma unroll
        for (int v = 0; v < 8; ++v) {
            int m  = v + 8 * g;
            int gm = Mbase + m;
            float sp = Sf[m * 256 + n];
            float pv = tanhf(acc[t][v] + ppsv);
            float hv = (sp + pstv * pv) * gate_in[(size_t)gm * DS_ + n] * gst;
            h_bf[(size_t)gm * DS_ + n] = (__bf16)hv;
            if ((gm & (T_ - 1)) == (T_ - 1)) out_tail[bi * DS_ + n] = hv;
        }
    }
}

// ---------------------------------------------------------------------------
// stageC: out = h @ W_out^T.  Block = 32 tokens x 1024 cols, 512 threads.
// h tile (K=256 complete) staged once -> zero barriers in K loop; B reused
// across 2 M-subtiles.
// ---------------------------------------------------------------------------
__global__ void __launch_bounds__(512) stageC(
    const __bf16* __restrict__ h_bf, const __bf16* __restrict__ Woutb,
    float* __restrict__ out)
{
    __shared__ __bf16 Hs[32 * 256];
    const int tid  = threadIdx.x;
    const int lane = tid & 31;
    const int wv   = tid >> 5;       // 0..15
    const int ln   = lane & 15;
    const int g    = lane >> 4;
    const int Mbase = blockIdx.x * 32;

    // cooperative load: 8192 bf16 = 512 threads x 16
    ((v16bf*)Hs)[tid] = ((const v16bf*)(h_bf + (size_t)Mbase * DS_))[tid];
    __syncthreads();

    v8f acc[2][4] = {};
    #pragma unroll 2
    for (int k0 = 0; k0 < DS_; k0 += 32) {
        v16bf bfr[4];
        #pragma unroll
        for (int t = 0; t < 4; ++t) {
            int n = (wv * 4 + t) * 16 + ln;
            bfr[t] = load_b_frag(Woutb, DS_, n, k0, lane);
        }
        v16bf a0 = load_a_frag<256>(&Hs[0],        lane, k0);
        v16bf a1 = load_a_frag<256>(&Hs[16 * 256], lane, k0);
        #pragma unroll
        for (int t = 0; t < 4; ++t) {
            acc[0][t] = wmma_bf16(a0, bfr[t], acc[0][t]);
            acc[1][t] = wmma_bf16(a1, bfr[t], acc[1][t]);
        }
    }
    #pragma unroll
    for (int ms = 0; ms < 2; ++ms) {
        #pragma unroll
        for (int t = 0; t < 4; ++t) {
            int n = (wv * 4 + t) * 16 + ln;
            #pragma unroll
            for (int v = 0; v < 8; ++v) {
                int m = ms * 16 + v + 8 * g;
                out[(size_t)(Mbase + m) * DM_ + n] = acc[ms][t][v];
            }
        }
    }
}

// ---------------------------------------------------------------------------
extern "C" void kernel_launch(void* const* d_in, const int* in_sizes, int n_in,
                              void* d_out, int out_size, void* d_ws, size_t ws_size,
                              hipStream_t stream)
{
    (void)in_sizes; (void)n_in; (void)out_size; (void)ws_size;
    const float* x     = (const float*)d_in[0];
    const float* prev  = (const float*)d_in[1];
    const float* W_in  = (const float*)d_in[2];
    const float* W_x   = (const float*)d_in[3];
    const float* W_s   = (const float*)d_in[4];
    const float* W_out = (const float*)d_in[5];
    const float* W_gx  = (const float*)d_in[6];
    const float* b_gx  = (const float*)d_in[7];
    const float* W_gs  = (const float*)d_in[8];
    const float* b_gs  = (const float*)d_in[9];
    const float* W_dc  = (const float*)d_in[10];
    const float* b_dc  = (const float*)d_in[11];
    const float* W_pp  = (const float*)d_in[12];
    const float* b_pp  = (const float*)d_in[13];
    const float* W_pg  = (const float*)d_in[14];
    const float* b_pg  = (const float*)d_in[15];

    char* ws = (char*)d_ws;
    size_t off = 0;
    auto alloc = [&](size_t bytes) -> char* {
        char* p = ws + off;
        off += (bytes + 255) & ~(size_t)255;
        return p;
    };
    float*  sWs     = (float*)alloc(B_ * DS_ * 4);
    float*  dcs     = (float*)alloc(B_ * DS_ * 4);
    float*  pps     = (float*)alloc(B_ * DS_ * 4);
    float*  pstr    = (float*)alloc(B_ * DS_ * 4);
    float*  gstate  = (float*)alloc(B_ * DS_ * 4);
    __bf16* Wcat    = (__bf16*)alloc((size_t)768 * 1024 * 2);
    __bf16* Wppa    = (__bf16*)alloc((size_t)256 * 256 * 2);
    __bf16* Woutb   = (__bf16*)alloc((size_t)1024 * 256 * 2);
    float*  s_pred  = (float*)alloc((size_t)BT_ * DS_ * 4);
    float*  gate_in = (float*)alloc((size_t)BT_ * DS_ * 4);
    __bf16* h_bf    = (__bf16*)alloc((size_t)BT_ * DS_ * 2);

    float* out      = (float*)d_out;
    float* out_tail = out + (size_t)BT_ * DM_;

    prepW<<<4352, 256, 0, stream>>>(W_in, W_x, W_gx, W_dc, W_pp, W_out, Wcat, Wppa, Woutb);
    prepS<<<40, 256, 0, stream>>>(prev, W_s, W_dc, b_dc, W_pp, b_pp, W_pg, b_pg, W_gs, b_gs,
                                  sWs, dcs, pps, pstr, gstate);
    stageA<<<BT_ / 32, 512, 0, stream>>>(x, prev, Wcat, b_gx, sWs, dcs, s_pred, gate_in);
    stageB<<<BT_ / 16, 256, 0, stream>>>(s_pred, gate_in, Wppa, pps, pstr, gstate, h_bf, out_tail);
    stageC<<<BT_ / 32, 512, 0, stream>>>(h_bf, Woutb, out);
}
